// GRUmodel_77180562309145
// MI455X (gfx1250) — matrix operand order repaired and checked by reading, not compiled, not run on hardware
//
#include <hip/hip_runtime.h>
#include <hip/hip_bf16.h>

typedef __attribute__((ext_vector_type(16))) _Float16 v16h;
typedef __attribute__((ext_vector_type(8)))  _Float16 v8h;
typedef __attribute__((ext_vector_type(8)))  float    v8f;

#define HID    128
#define NG     384          // 3*HID gate width
#define NB     32           // batch
#define TLEN   160
#define WS     8
#define NSTEP  144          // TLEN - 2*WS
#define SEQOUT 152          // WS + NSTEP
#define OUTC   2
#define NTHR   256

// ---- LDS layout (bytes). Total 326432 <= 320KB WGP LDS ----
#define OFF_W0    0                       // Whh0 f16  384x128
#define OFF_W1I   98304                   // Wih1 f16  384x128
#define OFF_W1H   196608                  // Whh1 f16  384x128
#define OFF_WIH0  294912                  // Wih0 f16  384x3
#define OFF_BIH0  297216                  // f32 384
#define OFF_BHH0  298752
#define OFF_BIH1  300288
#define OFF_BHH1  301824
#define OFF_WLIN  303360                  // f32 2x128
#define OFF_BLIN  304384                  // f32 2 (padded 32B)
#define OFF_H0    304416                  // f16 32x128
#define OFF_H1    312608                  // f16 32x128
#define OFF_XBUF  320800                  // f32 8x32x3 window
#define OFF_BUF   323872                  // f32 32x8x2 rolling outputs
#define OFF_PR    325920                  // f32 32x2 head output
#define OFF_NEW   326176                  // f32 32x2 scan "new"
#define LDS_BYTES 326432

__device__ __forceinline__ float sigm(float x) { return 1.0f / (1.0f + __expf(-x)); }

// B fragment (32x16 f16, K x N): lane col = lane%16, lanes<16 hold K 0..15,
// lanes>=16 hold K 16..31 -> 16 contiguous halves of weight row n.
__device__ __forceinline__ v16h load_bfrag(const _Float16* W, int n, int kc, int lane) {
    const _Float16* p = W + n * HID + kc * 32 + ((lane & 16) ? 16 : 0);
    return *(const v16h*)p;
}

// A fragment (16x32 f16, M x K): lane row = lane%16 (+16*mi), lanes<16 hold
// K 0..15, lanes>=16 hold K 16..31 -> two 8-half runs.
__device__ __forceinline__ v16h load_afrag(const _Float16* Hb, int mi, int kc, int lane) {
    int m = (lane & 15) + (mi << 4);
    const _Float16* p = Hb + m * HID + kc * 32 + ((lane & 16) ? 16 : 0);
    union { v16h v; v8h h[2]; } u;
    u.h[0] = *(const v8h*)p;
    u.h[1] = *(const v8h*)(p + 8);
    return u.v;
}

#define WMMA(a, b, c) \
    __builtin_amdgcn_wmma_f32_16x16x32_f16(false, (a), false, (b), (short)0, (c), false, false)

// One pred_last(): zero hidden, 8 GRU steps x 2 layers, linear head into sPr.
// Must be entered by all 256 threads (contains barriers; WMMA needs full EXEC).
__device__ void run_window(int tid) {
    extern __shared__ char smem[];
    _Float16* sW0   = (_Float16*)(smem + OFF_W0);
    _Float16* sW1i  = (_Float16*)(smem + OFF_W1I);
    _Float16* sW1h  = (_Float16*)(smem + OFF_W1H);
    _Float16* sWih0 = (_Float16*)(smem + OFF_WIH0);
    float*    sBih0 = (float*)(smem + OFF_BIH0);
    float*    sBhh0 = (float*)(smem + OFF_BHH0);
    float*    sBih1 = (float*)(smem + OFF_BIH1);
    float*    sBhh1 = (float*)(smem + OFF_BHH1);
    float*    sWlin = (float*)(smem + OFF_WLIN);
    float*    sBlin = (float*)(smem + OFF_BLIN);
    _Float16* sH0   = (_Float16*)(smem + OFF_H0);
    _Float16* sH1   = (_Float16*)(smem + OFF_H1);
    float*    sX    = (float*)(smem + OFF_XBUF);
    float*    sPr   = (float*)(smem + OFF_PR);

    const int lane = tid & 31;
    const int wv   = tid >> 5;          // wave id 0..7 -> hidden cols [wv*16, wv*16+16)
    const int j    = (wv << 4) + (lane & 15);
    const int mhi  = (lane >> 4) << 3;  // 0 or 8 (C-layout M offset per half-wave)

    // h0 = h1 = 0
    for (int i = tid; i < NB * HID; i += NTHR) {
        sH0[i] = (_Float16)0.0f;
        sH1[i] = (_Float16)0.0f;
    }
    __syncthreads();

    for (int step = 0; step < WS; ++step) {
        const float* xs = sX + step * (NB * 3);

        // ---------------- layer 0: acc = h0 @ Whh0^T ----------------
        v8f acc[3][2];
#pragma unroll
        for (int g = 0; g < 3; ++g) {
            v8f z = {0.f, 0.f, 0.f, 0.f, 0.f, 0.f, 0.f, 0.f};
            acc[g][0] = z; acc[g][1] = z;
        }
#pragma unroll
        for (int kc = 0; kc < 4; ++kc) {
            v16h a0 = load_afrag(sH0, 0, kc, lane);
            v16h a1 = load_afrag(sH0, 1, kc, lane);
#pragma unroll
            for (int g = 0; g < 3; ++g) {
                v16h b = load_bfrag(sW0, g * HID + j, kc, lane);
                acc[g][0] = WMMA(a0, b, acc[g][0]);
                acc[g][1] = WMMA(a1, b, acc[g][1]);
            }
        }
        // gates in-register; gx (K=3) via VALU
        float hnew[2][8];
#pragma unroll
        for (int mi = 0; mi < 2; ++mi) {
#pragma unroll
            for (int vr = 0; vr < 8; ++vr) {
                int m  = vr + mhi + (mi << 4);
                float x0 = xs[m * 3 + 0], x1 = xs[m * 3 + 1], x2 = xs[m * 3 + 2];
                int nr = j, nz = HID + j, nn = 2 * HID + j;
                float gxr = x0 * (float)sWih0[nr * 3] + x1 * (float)sWih0[nr * 3 + 1] + x2 * (float)sWih0[nr * 3 + 2];
                float gxz = x0 * (float)sWih0[nz * 3] + x1 * (float)sWih0[nz * 3 + 1] + x2 * (float)sWih0[nz * 3 + 2];
                float gxn = x0 * (float)sWih0[nn * 3] + x1 * (float)sWih0[nn * 3 + 1] + x2 * (float)sWih0[nn * 3 + 2];
                float r  = sigm(gxr + sBih0[nr] + acc[0][mi][vr] + sBhh0[nr]);
                float z  = sigm(gxz + sBih0[nz] + acc[1][mi][vr] + sBhh0[nz]);
                float hn = acc[2][mi][vr] + sBhh0[nn];
                float ng = tanhf(gxn + sBih0[nn] + r * hn);
                float ho = (float)sH0[m * HID + j];
                hnew[mi][vr] = (1.0f - z) * ng + z * ho;
            }
        }
        __syncthreads();   // all WMMA reads of h0 done
#pragma unroll
        for (int mi = 0; mi < 2; ++mi)
#pragma unroll
            for (int vr = 0; vr < 8; ++vr)
                sH0[(vr + mhi + (mi << 4)) * HID + j] = (_Float16)hnew[mi][vr];
        __syncthreads();   // y0_t = h0 ready

        // ------- layer 1: aX = y0 @ Wih1^T ; aH = h1 @ Whh1^T -------
        v8f aX[3][2], aH[3][2];
#pragma unroll
        for (int g = 0; g < 3; ++g) {
            v8f z = {0.f, 0.f, 0.f, 0.f, 0.f, 0.f, 0.f, 0.f};
            aX[g][0] = z; aX[g][1] = z; aH[g][0] = z; aH[g][1] = z;
        }
#pragma unroll
        for (int kc = 0; kc < 4; ++kc) {
            v16h ax0 = load_afrag(sH0, 0, kc, lane);
            v16h ax1 = load_afrag(sH0, 1, kc, lane);
            v16h ah0 = load_afrag(sH1, 0, kc, lane);
            v16h ah1 = load_afrag(sH1, 1, kc, lane);
#pragma unroll
            for (int g = 0; g < 3; ++g) {
                v16h bi = load_bfrag(sW1i, g * HID + j, kc, lane);
                v16h bh = load_bfrag(sW1h, g * HID + j, kc, lane);
                aX[g][0] = WMMA(ax0, bi, aX[g][0]);
                aX[g][1] = WMMA(ax1, bi, aX[g][1]);
                aH[g][0] = WMMA(ah0, bh, aH[g][0]);
                aH[g][1] = WMMA(ah1, bh, aH[g][1]);
            }
        }
#pragma unroll
        for (int mi = 0; mi < 2; ++mi) {
#pragma unroll
            for (int vr = 0; vr < 8; ++vr) {
                int m  = vr + mhi + (mi << 4);
                int nr = j, nz = HID + j, nn = 2 * HID + j;
                float r  = sigm(aX[0][mi][vr] + sBih1[nr] + aH[0][mi][vr] + sBhh1[nr]);
                float z  = sigm(aX[1][mi][vr] + sBih1[nz] + aH[1][mi][vr] + sBhh1[nz]);
                float hn = aH[2][mi][vr] + sBhh1[nn];
                float ng = tanhf(aX[2][mi][vr] + sBih1[nn] + r * hn);
                float ho = (float)sH1[m * HID + j];
                hnew[mi][vr] = (1.0f - z) * ng + z * ho;
            }
        }
        __syncthreads();   // all WMMA reads of h1 done
#pragma unroll
        for (int mi = 0; mi < 2; ++mi)
#pragma unroll
            for (int vr = 0; vr < 8; ++vr)
                sH1[(vr + mhi + (mi << 4)) * HID + j] = (_Float16)hnew[mi][vr];
        __syncthreads();
    }

    // linear head: pr = h1 @ Wlin^T + blin  (32x2)
    if (tid < NB * OUTC) {
        int m = tid >> 1, o = tid & 1;
        float s = sBlin[o];
        for (int k = 0; k < HID; ++k)
            s += (float)sH1[m * HID + k] * sWlin[o * HID + k];
        sPr[m * OUTC + o] = s;
    }
    __syncthreads();
}

__global__ void __launch_bounds__(NTHR)
gru_rollout_kernel(const float* __restrict__ traj,
                   const float* __restrict__ Wih0, const float* __restrict__ Whh0,
                   const float* __restrict__ bih0, const float* __restrict__ bhh0,
                   const float* __restrict__ Wih1, const float* __restrict__ Whh1,
                   const float* __restrict__ bih1, const float* __restrict__ bhh1,
                   const float* __restrict__ Wlin, const float* __restrict__ blin,
                   float* __restrict__ out) {
    extern __shared__ char smem[];
    _Float16* sW0   = (_Float16*)(smem + OFF_W0);
    _Float16* sW1i  = (_Float16*)(smem + OFF_W1I);
    _Float16* sW1h  = (_Float16*)(smem + OFF_W1H);
    _Float16* sWih0 = (_Float16*)(smem + OFF_WIH0);
    float*    sBih0 = (float*)(smem + OFF_BIH0);
    float*    sBhh0 = (float*)(smem + OFF_BHH0);
    float*    sBih1 = (float*)(smem + OFF_BIH1);
    float*    sBhh1 = (float*)(smem + OFF_BHH1);
    float*    sWlin = (float*)(smem + OFF_WLIN);
    float*    sBlin = (float*)(smem + OFF_BLIN);
    _Float16* sH0   = (_Float16*)(smem + OFF_H0);
    _Float16* sH1   = (_Float16*)(smem + OFF_H1);
    float*    sX    = (float*)(smem + OFF_XBUF);
    float*    sBuf  = (float*)(smem + OFF_BUF);   // [b][q][c] = b*16+q*2+c
    float*    sPr   = (float*)(smem + OFF_PR);
    float*    sNew  = (float*)(smem + OFF_NEW);

    const int tid = threadIdx.x;

    // ---- one-time: stage all weights into LDS (f16 for WMMA operands) ----
    for (int i = tid; i < NG * HID; i += NTHR) {
        sW0[i]  = (_Float16)Whh0[i];
        sW1i[i] = (_Float16)Wih1[i];
        sW1h[i] = (_Float16)Whh1[i];
    }
    for (int i = tid; i < NG * 3; i += NTHR) sWih0[i] = (_Float16)Wih0[i];
    for (int i = tid; i < NG; i += NTHR) {
        sBih0[i] = bih0[i]; sBhh0[i] = bhh0[i];
        sBih1[i] = bih1[i]; sBhh1[i] = bhh1[i];
    }
    for (int i = tid; i < OUTC * HID; i += NTHR) sWlin[i] = Wlin[i];
    if (tid < OUTC) sBlin[tid] = blin[tid];
    __syncthreads();

    // ---------------- Phase A: 8 init windows ----------------
    for (int t = 0; t < WS; ++t) {
        for (int i = tid; i < WS * NB * 3; i += NTHR) {
            int s = i / (NB * 3), r = i % (NB * 3), b = r / 3, d = r % 3;
            float v;
            if (t == 0) {
                v = traj[b * TLEN * 3 + s * 3 + d];
            } else if (s < WS - 1) {
                v = traj[b * TLEN * 3 + (t + s) * 3 + d];
            } else if (d == 0) {
                v = traj[b * TLEN * 3 + (WS + t) * 3];
            } else {
                v = sBuf[b * 16 + (t - 1) * 2 + (d - 1)];
            }
            sX[i] = v;
        }
        __syncthreads();
        run_window(tid);
        if (tid < NB * OUTC) {
            int b = tid >> 1, c = tid & 1;
            float v = traj[b * TLEN * 3 + (WS + t - 1) * 3 + (c + 1)] + sPr[b * OUTC + c];
            sBuf[b * 16 + t * 2 + c]            = v;   // out_init feeds the scan buffer
            out[b * SEQOUT * OUTC + t * OUTC + c] = v;
        }
        __syncthreads();
    }

    // ---------------- Phase B: 144 scan steps ----------------
    for (int s = 0; s < NSTEP; ++s) {
        for (int i = tid; i < WS * NB * 3; i += NTHR) {
            int q = i / (NB * 3), r = i % (NB * 3), b = r / 3, d = r % 3;
            sX[i] = (d == 0) ? traj[b * TLEN * 3 + (WS + s + q) * 3]
                             : sBuf[b * 16 + q * 2 + (d - 1)];
        }
        __syncthreads();
        run_window(tid);
        if (tid < NB * OUTC) {
            int b = tid >> 1, c = tid & 1;
            float v = sBuf[b * 16 + 7 * 2 + c] + sPr[b * OUTC + c];
            sNew[b * OUTC + c] = v;
            out[b * SEQOUT * OUTC + (WS + s) * OUTC + c] = v;
        }
        __syncthreads();
        // shift-append buf (read-all then write-all across a barrier)
        int i0 = tid, i1 = tid + NTHR;   // 512 elements total
        int b0 = i0 >> 4, rq0 = i0 & 15, q0 = rq0 >> 1, c0 = rq0 & 1;
        int b1 = i1 >> 4, rq1 = i1 & 15, q1 = rq1 >> 1, c1 = rq1 & 1;
        float t0 = (q0 < 7) ? sBuf[b0 * 16 + (q0 + 1) * 2 + c0] : sNew[b0 * 2 + c0];
        float t1 = (q1 < 7) ? sBuf[b1 * 16 + (q1 + 1) * 2 + c1] : sNew[b1 * 2 + c1];
        __syncthreads();
        sBuf[i0] = t0;
        sBuf[i1] = t1;
        __syncthreads();
    }

    // hidden output = stack([h0_T, h1_T]) from the final scan window
    const int HOFF = NB * SEQOUT * OUTC;   // 9728
    for (int i = tid; i < NB * HID; i += NTHR) {
        out[HOFF + i]            = (float)sH0[i];
        out[HOFF + NB * HID + i] = (float)sH1[i];
    }
}

extern "C" void kernel_launch(void* const* d_in, const int* in_sizes, int n_in,
                              void* d_out, int out_size, void* d_ws, size_t ws_size,
                              hipStream_t stream) {
    (void)in_sizes; (void)n_in; (void)out_size; (void)d_ws; (void)ws_size;
    gru_rollout_kernel<<<dim3(1), dim3(NTHR), LDS_BYTES, stream>>>(
        (const float*)d_in[0],  // traj
        (const float*)d_in[1],  // Wih0
        (const float*)d_in[2],  // Whh0
        (const float*)d_in[3],  // bih0
        (const float*)d_in[4],  // bhh0
        (const float*)d_in[5],  // Wih1
        (const float*)d_in[6],  // Whh1
        (const float*)d_in[7],  // bih1
        (const float*)d_in[8],  // bhh1
        (const float*)d_in[9],  // Wlin
        (const float*)d_in[10], // blin
        (float*)d_out);
}
